// FusedTransformerBlock_8787503088279
// MI455X (gfx1250) — matrix operand order repaired and checked by reading, not compiled, not run on hardware
//
#include <hip/hip_runtime.h>
#include <hip/hip_bf16.h>
#include <math.h>

#define SDIM 2048
#define BDIM 2
#define HDIM 1024
#define NHEAD 16
#define HEADD 64
#define NTOK (SDIM * BDIM)          // 4096 tokens
#define H3 (3 * HDIM)               // 3072
#define ROWSTRIDE (BDIM * H3)       // 6144 elements between s and s+1 in qkv

typedef __bf16 bf16_t;
typedef __bf16 v16bf __attribute__((ext_vector_type(16)));
typedef __bf16 v8bf  __attribute__((ext_vector_type(8)));
typedef float  v8f   __attribute__((ext_vector_type(8)));

// ---------------------------------------------------------------------------
// Fragment loaders for wave32 WMMA (16-bit operands), contiguous k order used by both A and B.
// A 16x32 (MxK): lanes 0-15 row=lane, elems 0..15 -> K 0..15
//                lanes 16-31 row=lane-16, elems -> K 16..31.
// B 32x16 (KxN): lanes 0-15 col=lane, elems 0..15 -> K 0..15
//                lanes 16-31 col=lane-16, elems -> K 16..31.
// ---------------------------------------------------------------------------
__device__ __forceinline__ v16bf load_frag_a(const bf16_t* base, int ld) {
    int lane = threadIdx.x & 31;
    const bf16_t* p = base + (size_t)(lane & 15) * ld + ((lane >> 4) << 4);
    v8bf lo = *(const v8bf*)(p);
    v8bf hi = *(const v8bf*)(p + 8);
    v16bf f;
#pragma unroll
    for (int i = 0; i < 8; ++i) { f[i] = lo[i]; f[i + 8] = hi[i]; }
    return f;
}

__device__ __forceinline__ v16bf load_frag_b(const bf16_t* base, int ld) {
    int lane = threadIdx.x & 31;
    const bf16_t* p = base + (size_t)(lane & 15) * ld + ((lane >> 4) << 4);
    v8bf lo = *(const v8bf*)(p);
    v8bf hi = *(const v8bf*)(p + 8);
    v16bf f;
#pragma unroll
    for (int i = 0; i < 8; ++i) { f[i] = lo[i]; f[i + 8] = hi[i]; }
    return f;
}

// B operand from a column-major view: B[k][n] = base[k*ld + n]  (used for P @ V)
__device__ __forceinline__ v16bf load_frag_b_cm(const bf16_t* base, int ld) {
    int lane = threadIdx.x & 31;
    const bf16_t* p = base + (lane & 15) + (size_t)((lane >> 4) << 4) * ld;
    v16bf f;
#pragma unroll
    for (int j = 0; j < 16; ++j) f[j] = p[(size_t)j * ld];
    return f;
}

__device__ __forceinline__ v8f wmma_bf16(v16bf a, v16bf b, v8f c) {
    return __builtin_amdgcn_wmma_f32_16x16x32_bf16(false, a, false, b, (short)0, c,
                                                   false, false);
}

__device__ __forceinline__ float gelu_tanh(float v) {
    const float c = 0.7978845608028654f;
    float t = tanhf(c * (v + 0.044715f * v * v * v));
    return 0.5f * v * (1.0f + t);
}

// ---------------------------------------------------------------------------
// fp32 -> bf16 convert (weights / staging)
// ---------------------------------------------------------------------------
__global__ void cvt_bf16_kernel(const float* __restrict__ in, bf16_t* __restrict__ out,
                                int n) {
    int i = blockIdx.x * blockDim.x + threadIdx.x;
    if (i < n) out[i] = (bf16_t)in[i];
}

// ---------------------------------------------------------------------------
// LayerNorm over H=1024, one token per block (256 threads, 4 elems/thread),
// writes bf16 normalized activations.
// ---------------------------------------------------------------------------
__global__ __launch_bounds__(256) void ln_bf16_kernel(const float* __restrict__ x,
                                                      const float* __restrict__ w,
                                                      const float* __restrict__ b,
                                                      bf16_t* __restrict__ out) {
    __shared__ float red[256];
    int t = blockIdx.x;
    int tid = threadIdx.x;
    const float4* xr = (const float4*)(x + (size_t)t * HDIM);
    float4 v = xr[tid];
    float s = v.x + v.y + v.z + v.w;
    red[tid] = s;
    __syncthreads();
#pragma unroll
    for (int off = 128; off > 0; off >>= 1) {
        if (tid < off) red[tid] += red[tid + off];
        __syncthreads();
    }
    float mean = red[0] * (1.0f / HDIM);
    __syncthreads();
    float d0 = v.x - mean, d1 = v.y - mean, d2 = v.z - mean, d3 = v.w - mean;
    red[tid] = d0 * d0 + d1 * d1 + d2 * d2 + d3 * d3;
    __syncthreads();
#pragma unroll
    for (int off = 128; off > 0; off >>= 1) {
        if (tid < off) red[tid] += red[tid + off];
        __syncthreads();
    }
    float rstd = rsqrtf(red[0] * (1.0f / HDIM) + 1e-5f);
    int c = tid * 4;
    bf16_t* o = out + (size_t)t * HDIM + c;
    o[0] = (bf16_t)(d0 * rstd * w[c + 0] + b[c + 0]);
    o[1] = (bf16_t)(d1 * rstd * w[c + 1] + b[c + 1]);
    o[2] = (bf16_t)(d2 * rstd * w[c + 2] + b[c + 2]);
    o[3] = (bf16_t)(d3 * rstd * w[c + 3] + b[c + 3]);
}

// ---------------------------------------------------------------------------
// WMMA GEMM: out[M,N] = A[M,K] @ W[N,K]^T + bias, with fused epilogues.
// Block: 256 threads = 8 waves; block tile 128x128; wave tile 32(M) x 64(N).
// mode 0: f32 out (+ optional residual)   mode 1: bf16(gelu)   mode 2: bf16
// ---------------------------------------------------------------------------
__global__ __launch_bounds__(256) void wmma_gemm_kernel(
    const bf16_t* __restrict__ A, int lda, const bf16_t* __restrict__ W, int ldw,
    const float* __restrict__ bias, const float* __restrict__ residual,
    float* __restrict__ outf, bf16_t* __restrict__ outh, int M, int N, int K,
    int mode) {
    int wave = threadIdx.x >> 5;
    int lane = threadIdx.x & 31;
    int wm = blockIdx.y * 128 + (wave & 3) * 32;
    int wn = blockIdx.x * 128 + (wave >> 2) * 64;

    v8f acc[2][4] = {};
    for (int k0 = 0; k0 < K; k0 += 32) {
        v16bf a0 = load_frag_a(A + (size_t)wm * lda + k0, lda);
        v16bf a1 = load_frag_a(A + (size_t)(wm + 16) * lda + k0, lda);
        v16bf bfr[4];
#pragma unroll
        for (int j = 0; j < 4; ++j)
            bfr[j] = load_frag_b(W + (size_t)(wn + 16 * j) * ldw + k0, ldw);
#pragma unroll
        for (int j = 0; j < 4; ++j) {
            acc[0][j] = wmma_bf16(a0, bfr[j], acc[0][j]);
            acc[1][j] = wmma_bf16(a1, bfr[j], acc[1][j]);
        }
    }

    int cidx = lane & 15;
    int rhi = (lane >> 4) * 8;
#pragma unroll
    for (int mi = 0; mi < 2; ++mi) {
#pragma unroll
        for (int j = 0; j < 4; ++j) {
            int col = wn + 16 * j + cidx;
            float bv = bias[col];
#pragma unroll
            for (int r = 0; r < 8; ++r) {
                int row = wm + 16 * mi + rhi + r;
                size_t idx = (size_t)row * N + col;
                float v = acc[mi][j][r] + bv;
                if (mode == 0) {
                    if (residual) v += residual[idx];
                    outf[idx] = v;
                } else if (mode == 1) {
                    outh[idx] = (bf16_t)gelu_tanh(v);
                } else {
                    outh[idx] = (bf16_t)v;
                }
            }
        }
    }
}

// ---------------------------------------------------------------------------
// Flash-style causal attention. qkv: [token = s*B+b, 3H] bf16, head h occupies
// columns h*192 + {0:q, 64:k, 128:v}. Grid: (S/128, B*NH). 8 waves/block, each
// wave owns 16 query rows of the 128-row q tile. Output bf16 [token, H] in
// [s, b, h*64+d] order (matches the reference transpose back to [S,B,H]).
// ---------------------------------------------------------------------------
__global__ __launch_bounds__(256) void attn_kernel(const bf16_t* __restrict__ qkv,
                                                   bf16_t* __restrict__ out) {
    int bh = blockIdx.y;
    int b = bh / NHEAD;
    int h = bh % NHEAD;
    int wave = threadIdx.x >> 5;
    int lane = threadIdx.x & 31;
    int q0 = blockIdx.x * 128 + wave * 16;  // this wave's first query row (s index)
    int cidx = lane & 15;
    int rhi = (lane >> 4) * 8;

    const bf16_t* qbase = qkv + (size_t)b * H3 + h * (3 * HEADD);
    const bf16_t* kbase = qbase + HEADD;
    const bf16_t* vbase = qbase + 2 * HEADD;

    __shared__ __align__(16) bf16_t Plds[8][16 * 128];
    bf16_t* pw = &Plds[wave][0];

    // Q fragments for this wave's 16 rows (K = head dim 64 -> two 32-chunks)
    v16bf qfrag[2];
#pragma unroll
    for (int kk = 0; kk < 2; ++kk)
        qfrag[kk] = load_frag_a(qbase + (size_t)q0 * ROWSTRIDE + kk * 32, ROWSTRIDE);

    float m_i[8], l_i[8];
#pragma unroll
    for (int r = 0; r < 8; ++r) { m_i[r] = -INFINITY; l_i[r] = 0.0f; }
    v8f o_acc[4] = {};

    for (int kb = 0; kb < q0 + 16; kb += 128) {
        // ---- scores S[16q,128k] = Q @ K^T, scaled + causal mask ----
        v8f sacc[8] = {};
#pragma unroll
        for (int t = 0; t < 8; ++t) {
            const bf16_t* kp = kbase + (size_t)(kb + 16 * t) * ROWSTRIDE;
#pragma unroll
            for (int kk = 0; kk < 2; ++kk) {
                v16bf kf = load_frag_b(kp + kk * 32, ROWSTRIDE);
                sacc[t] = wmma_bf16(qfrag[kk], kf, sacc[t]);
            }
        }
        float tmax[8];
#pragma unroll
        for (int r = 0; r < 8; ++r) tmax[r] = -1e30f;
#pragma unroll
        for (int t = 0; t < 8; ++t) {
            int key = kb + 16 * t + cidx;
#pragma unroll
            for (int r = 0; r < 8; ++r) {
                float s = sacc[t][r] * 0.125f;  // 1/sqrt(64)
                if (key > q0 + rhi + r) s = -1e30f;
                sacc[t][r] = s;
                tmax[r] = fmaxf(tmax[r], s);
            }
        }
        // row max across the 16 lane-columns (stays inside each half-group)
#pragma unroll
        for (int off = 1; off < 16; off <<= 1)
#pragma unroll
            for (int r = 0; r < 8; ++r)
                tmax[r] = fmaxf(tmax[r], __shfl_xor(tmax[r], off, 32));

        float alpha[8], rowsum[8];
#pragma unroll
        for (int r = 0; r < 8; ++r) {
            float mnew = fmaxf(m_i[r], tmax[r]);
            alpha[r] = __expf(m_i[r] - mnew);
            m_i[r] = mnew;
            rowsum[r] = 0.0f;
        }
#pragma unroll
        for (int t = 0; t < 8; ++t)
#pragma unroll
            for (int r = 0; r < 8; ++r) {
                float p = __expf(sacc[t][r] - m_i[r]);
                sacc[t][r] = p;
                rowsum[r] += p;
            }
#pragma unroll
        for (int off = 1; off < 16; off <<= 1)
#pragma unroll
            for (int r = 0; r < 8; ++r)
                rowsum[r] += __shfl_xor(rowsum[r], off, 32);
#pragma unroll
        for (int r = 0; r < 8; ++r) l_i[r] = l_i[r] * alpha[r] + rowsum[r];
#pragma unroll
        for (int u = 0; u < 4; ++u)
#pragma unroll
            for (int r = 0; r < 8; ++r) o_acc[u][r] *= alpha[r];

        // ---- stage P (bf16) into this wave's LDS tile, row-major [16,128] ----
#pragma unroll
        for (int t = 0; t < 8; ++t)
#pragma unroll
            for (int r = 0; r < 8; ++r)
                pw[(rhi + r) * 128 + 16 * t + cidx] = (bf16_t)sacc[t][r];

        // ---- O += P @ V ----
#pragma unroll
        for (int kk = 0; kk < 4; ++kk) {
            v16bf pf = load_frag_a(pw + kk * 32, 128);
            const bf16_t* vp = vbase + (size_t)(kb + kk * 32) * ROWSTRIDE;
#pragma unroll
            for (int u = 0; u < 4; ++u) {
                v16bf vf = load_frag_b_cm(vp + 16 * u, ROWSTRIDE);
                o_acc[u] = wmma_bf16(pf, vf, o_acc[u]);
            }
        }
    }

    float inv_l[8];
#pragma unroll
    for (int r = 0; r < 8; ++r) inv_l[r] = 1.0f / l_i[r];
#pragma unroll
    for (int u = 0; u < 4; ++u) {
        int d = 16 * u + cidx;
#pragma unroll
        for (int r = 0; r < 8; ++r) {
            int s = q0 + rhi + r;
            out[((size_t)s * BDIM + b) * HDIM + h * HEADD + d] =
                (bf16_t)(o_acc[u][r] * inv_l[r]);
        }
    }
}

// ---------------------------------------------------------------------------
// Host launcher
// ---------------------------------------------------------------------------
extern "C" void kernel_launch(void* const* d_in, const int* in_sizes, int n_in,
                              void* d_out, int out_size, void* d_ws, size_t ws_size,
                              hipStream_t stream) {
    const float* x      = (const float*)d_in[0];
    const float* ln1_w  = (const float*)d_in[1];
    const float* ln1_b  = (const float*)d_in[2];
    const float* w_qkv  = (const float*)d_in[3];
    const float* b_qkv  = (const float*)d_in[4];
    const float* w_proj = (const float*)d_in[5];
    const float* b_proj = (const float*)d_in[6];
    const float* ln2_w  = (const float*)d_in[7];
    const float* ln2_b  = (const float*)d_in[8];
    const float* w_fc1  = (const float*)d_in[9];
    const float* b_fc1  = (const float*)d_in[10];
    const float* w_fc2  = (const float*)d_in[11];
    const float* b_fc2  = (const float*)d_in[12];
    float* out = (float*)d_out;

    char* ws = (char*)d_ws;
    auto carve = [&](size_t bytes) {
        char* p = ws;
        ws += (bytes + 255) & ~(size_t)255;
        return p;
    };
    bf16_t* wb_qkv  = (bf16_t*)carve((size_t)H3 * HDIM * 2);
    bf16_t* wb_proj = (bf16_t*)carve((size_t)HDIM * HDIM * 2);
    bf16_t* wb_fc1  = (bf16_t*)carve((size_t)4 * HDIM * HDIM * 2);
    bf16_t* wb_fc2  = (bf16_t*)carve((size_t)4 * HDIM * HDIM * 2);
    bf16_t* lnx     = (bf16_t*)carve((size_t)NTOK * HDIM * 2);
    bf16_t* qkvb    = (bf16_t*)carve((size_t)NTOK * H3 * 2);
    bf16_t* attnb   = (bf16_t*)carve((size_t)NTOK * HDIM * 2);
    float*  x1      = (float*) carve((size_t)NTOK * HDIM * 4);
    bf16_t* ln2x    = (bf16_t*)carve((size_t)NTOK * HDIM * 2);
    bf16_t* hmid    = (bf16_t*)carve((size_t)NTOK * 4 * HDIM * 2);

    // 1) weights -> bf16
    {
        int n = H3 * HDIM;
        cvt_bf16_kernel<<<(n + 255) / 256, 256, 0, stream>>>(w_qkv, wb_qkv, n);
        n = HDIM * HDIM;
        cvt_bf16_kernel<<<(n + 255) / 256, 256, 0, stream>>>(w_proj, wb_proj, n);
        n = 4 * HDIM * HDIM;
        cvt_bf16_kernel<<<(n + 255) / 256, 256, 0, stream>>>(w_fc1, wb_fc1, n);
        cvt_bf16_kernel<<<(n + 255) / 256, 256, 0, stream>>>(w_fc2, wb_fc2, n);
    }
    // 2) LN1
    ln_bf16_kernel<<<NTOK, 256, 0, stream>>>(x, ln1_w, ln1_b, lnx);
    // 3) QKV: [4096,3072] = lnx[4096,1024] @ w_qkv^T  -> bf16
    wmma_gemm_kernel<<<dim3(H3 / 128, NTOK / 128), 256, 0, stream>>>(
        lnx, HDIM, wb_qkv, HDIM, b_qkv, nullptr, nullptr, qkvb, NTOK, H3, HDIM, 2);
    // 4) causal attention
    attn_kernel<<<dim3(SDIM / 128, BDIM * NHEAD), 256, 0, stream>>>(qkvb, attnb);
    // 5) out-proj + residual(x) -> x1 (f32)
    wmma_gemm_kernel<<<dim3(HDIM / 128, NTOK / 128), 256, 0, stream>>>(
        attnb, HDIM, wb_proj, HDIM, b_proj, x, x1, nullptr, NTOK, HDIM, HDIM, 0);
    // 6) LN2
    ln_bf16_kernel<<<NTOK, 256, 0, stream>>>(x1, ln2_w, ln2_b, ln2x);
    // 7) FC1 + gelu -> bf16
    wmma_gemm_kernel<<<dim3(4 * HDIM / 128, NTOK / 128), 256, 0, stream>>>(
        ln2x, HDIM, wb_fc1, HDIM, b_fc1, nullptr, nullptr, hmid, NTOK, 4 * HDIM,
        HDIM, 1);
    // 8) FC2 + residual(x1) -> out (f32)
    wmma_gemm_kernel<<<dim3(HDIM / 128, NTOK / 128), 256, 0, stream>>>(
        hmid, 4 * HDIM, wb_fc2, 4 * HDIM, b_fc2, x1, out, nullptr, NTOK, HDIM,
        4 * HDIM, 0);
}
